// MultiHeadAttention_77077483094971
// MI455X (gfx1250) — matrix operand order repaired and checked by reading, not compiled, not run on hardware
//
#include <hip/hip_runtime.h>
#include <stdint.h>

// MHA for B=2, S=2048, D=1024, H=16, HD=64 on gfx1250 (wave32, WMMA bf16).
#define Bb 2
#define Ss 2048
#define Dd 1024
#define Hh 16
#define HDd 64
#define SCALEf 0.125f   // 1/sqrt(64)

typedef __attribute__((ext_vector_type(16))) __bf16 v16bf;
typedef __attribute__((ext_vector_type(8)))  float  v8f;

union BFrag { v16bf v; unsigned u[8]; };
union U4    { uint4 q; unsigned short h[8]; };

__device__ __forceinline__ unsigned short f2bf(float f) {
  unsigned u = __float_as_uint(f);
  u += 0x7FFFu + ((u >> 16) & 1u);          // round-to-nearest-even
  return (unsigned short)(u >> 16);
}

__device__ __forceinline__ v8f wmma_bf16(const BFrag& a, const BFrag& b, v8f c) {
  return __builtin_amdgcn_wmma_f32_16x16x32_bf16(false, a.v, false, b.v,
                                                 (short)0, c, false, false);
}

// ---------------------------------------------------------------------------
// GEMM: out[M,N] = bf16(A_f32[M,K]) @ bf16(W_f32[K,N]) + bias[N]
// M=4096, N=K=1024. Block tile 128x128, K-step 32. 8 waves: 4(M) x 2(N),
// each wave 32x64 -> 2x4 wmma accumulators. fp32->bf16 conversion happens in
// the LDS loaders (so no async copy here -- data is transformed in flight).
// OUT_F32: store fp32 (final projection) else bf16 (q/k/v).
// ---------------------------------------------------------------------------
template <bool OUT_F32>
__global__ __launch_bounds__(256) void gemm_bias_kernel(
    const float* __restrict__ A, const float* __restrict__ W,
    const float* __restrict__ bias, void* __restrict__ outp) {
  const int NN = 1024, KK = 1024;
  __shared__ unsigned short a_lds[128][32];   // [m][k] bf16
  __shared__ unsigned short w_lds[128][32];   // [n][k] bf16 (W transposed)

  const int tid  = threadIdx.x;
  const int lane = tid & 31;
  const int w8   = tid >> 5;
  const int wm   = w8 & 3;          // M quadrant (32 rows)
  const int wn   = w8 >> 2;         // N half (64 cols)
  const int hsel = lane >> 4;       // lane half
  const int l16  = lane & 15;
  const int m0   = blockIdx.y * 128;
  const int n0   = blockIdx.x * 128;

  v8f acc[2][4];
#pragma unroll
  for (int mt = 0; mt < 2; ++mt)
#pragma unroll
    for (int nt = 0; nt < 4; ++nt) acc[mt][nt] = 0.0f;

  for (int k0 = 0; k0 < KK; k0 += 32) {
    // A tile: 128x32 fp32 -> bf16
#pragma unroll
    for (int r = 0; r < 4; ++r) {
      int idx = tid + r * 256;                 // float4 units
      int row = idx >> 3;
      int c4  = (idx & 7) << 2;
      float4 f = *(const float4*)(A + (size_t)(m0 + row) * KK + k0 + c4);
      ushort4 h;
      h.x = f2bf(f.x); h.y = f2bf(f.y); h.z = f2bf(f.z); h.w = f2bf(f.w);
      *(ushort4*)&a_lds[row][c4] = h;
    }
    // W tile: 32x128 fp32 -> bf16, transposed to [n][k]
#pragma unroll
    for (int r = 0; r < 4; ++r) {
      int idx = tid + r * 256;
      int kk  = idx >> 5;
      int nn  = (idx & 31) << 2;
      float4 f = *(const float4*)(W + (size_t)(k0 + kk) * NN + n0 + nn);
      w_lds[nn + 0][kk] = f2bf(f.x);
      w_lds[nn + 1][kk] = f2bf(f.y);
      w_lds[nn + 2][kk] = f2bf(f.z);
      w_lds[nn + 3][kk] = f2bf(f.w);
    }
    __syncthreads();

    BFrag af[2];
#pragma unroll
    for (int mt = 0; mt < 2; ++mt) {
      int m = wm * 32 + mt * 16 + l16;
#pragma unroll
      for (int vv = 0; vv < 8; ++vv) {
        int kk = ((vv & 4) << 2) + hsel * 8 + ((vv & 3) << 1);
        af[mt].u[vv] = *(const unsigned*)&a_lds[m][kk];
      }
    }
    BFrag bf[4];
#pragma unroll
    for (int nt = 0; nt < 4; ++nt) {
      int n = wn * 64 + nt * 16 + l16;
#pragma unroll
      for (int vv = 0; vv < 8; ++vv) {
        int kk = ((vv & 4) << 2) + hsel * 8 + ((vv & 3) << 1);
        bf[nt].u[vv] = *(const unsigned*)&w_lds[n][kk];
      }
    }
#pragma unroll
    for (int mt = 0; mt < 2; ++mt)
#pragma unroll
      for (int nt = 0; nt < 4; ++nt)
        acc[mt][nt] = wmma_bf16(af[mt], bf[nt], acc[mt][nt]);
    __syncthreads();
  }

  // epilogue: C element (m = r + 8*hsel, n = l16) per accumulator
#pragma unroll
  for (int mt = 0; mt < 2; ++mt)
#pragma unroll
    for (int nt = 0; nt < 4; ++nt) {
      int gn = n0 + wn * 64 + nt * 16 + l16;
      float bval = bias[gn];
#pragma unroll
      for (int r = 0; r < 8; ++r) {
        int gm = m0 + wm * 32 + mt * 16 + r + 8 * hsel;
        float val = acc[mt][nt][r] + bval;
        if (OUT_F32)
          ((float*)outp)[(size_t)gm * NN + gn] = val;
        else
          ((unsigned short*)outp)[(size_t)gm * NN + gn] = f2bf(val);
      }
    }
}

// ---------------------------------------------------------------------------
// Attention with head-axis softmax.
// Grid: B * S/16 blocks; 256 threads = 8 waves; wave w owns heads 2w, 2w+1.
// Per 32-wide j-tile:
//   - K tile copied global->LDS with GLOBAL_LOAD_ASYNC_TO_LDS_B128 (ASYNCcnt),
//     V tile transposed through VGPRs (async can't transpose).
//   - scores (WMMA, K=hd=64), scale+mask, softmax over the 16 heads via LDS
//     [i][j][h], attn->bf16 LDS [h][i][j], attn@V (WMMA, K=j=32) into
//     persistent fp32 accumulators. Unnormalized sum over j (faithful to ref).
// ---------------------------------------------------------------------------
__global__ __launch_bounds__(256) void attn_kernel(
    const unsigned short* __restrict__ Q, const unsigned short* __restrict__ K,
    const unsigned short* __restrict__ V, const int* __restrict__ mask,
    float* __restrict__ ctx) {
  extern __shared__ char smem[];
  unsigned short* k_lds = (unsigned short*)smem;             // [32][1024]
  unsigned short* v_lds = k_lds + 32 * 1024;                 // [1024][32] (d-major)
  float*          sc    = (float*)(v_lds + 1024 * 32);       // [16 i][32 j][16 h]
  unsigned short* at    = (unsigned short*)(sc + 16 * 32 * 16); // [16 h][16 i][32 j]

  const int tid  = threadIdx.x;
  const int lane = tid & 31;
  const int w8   = tid >> 5;
  const int hsel = lane >> 4;
  const int l16  = lane & 15;
  const int b    = blockIdx.x >> 7;          // 128 i-tiles per batch
  const int i0   = (blockIdx.x & 127) << 4;

  // Q fragments (A-matrix, K=hd): [head][hd-half], persistent in registers
  BFrag qf[2][2];
#pragma unroll
  for (int hh = 0; hh < 2; ++hh) {
    int h = w8 * 2 + hh;
#pragma unroll
    for (int hf = 0; hf < 2; ++hf)
#pragma unroll
      for (int vv = 0; vv < 8; ++vv) {
        int hd = hf * 32 + ((vv & 4) << 2) + hsel * 8 + ((vv & 3) << 1);
        qf[hh][hf].u[vv] =
            *(const unsigned*)(Q + (size_t)(b * Ss + i0 + l16) * Dd + h * HDd + hd);
      }
  }

  v8f oacc[2][4];
#pragma unroll
  for (int hh = 0; hh < 2; ++hh)
#pragma unroll
    for (int nt = 0; nt < 4; ++nt) oacc[hh][nt] = 0.0f;

  for (int j0 = 0; j0 < Ss; j0 += 32) {
    // K tile: async copy to LDS (bf16 pass-through, ASYNCcnt-tracked).
    // V tile: transpose to [d][j] through VGPRs.
#pragma unroll
    for (int r = 0; r < 16; ++r) {
      int idx = tid + r * 256;                // uint4 units, 32 rows x 128
      int row = idx >> 7;
      int d8  = (idx & 127) << 3;
      size_t g = (size_t)(b * Ss + j0 + row) * Dd + d8;
      unsigned kaddr = (unsigned)(uintptr_t)(k_lds + row * 1024 + d8);
      asm volatile("global_load_async_to_lds_b128 %0, %1, off"
                   :: "v"(kaddr), "v"(K + g) : "memory");
      U4 s; s.q = *(const uint4*)(V + g);
#pragma unroll
      for (int t = 0; t < 8; ++t) v_lds[(d8 + t) * 32 + row] = s.h[t];
    }
    // warm L2 for the next j-tile (global_prefetch_b8)
    if (j0 + 32 < Ss) {
      size_t pg = (size_t)(b * Ss + j0 + 32 + (tid >> 3)) * Dd + ((tid & 7) << 7);
      __builtin_prefetch(K + pg, 0, 1);
      __builtin_prefetch(V + pg, 0, 1);
    }
    // async LDS writes are NOT covered by DScnt -> explicit wait, then barrier
    asm volatile("s_wait_asynccnt 0x0" ::: "memory");
    __syncthreads();

    // scores: S[h][i][j] = q_h (16x64) @ k_h^T (64x16), scaled + masked
#pragma unroll
    for (int hh = 0; hh < 2; ++hh) {
      int h = w8 * 2 + hh;
#pragma unroll
      for (int jt = 0; jt < 2; ++jt) {
        BFrag kf0, kf1;
        int jrow = jt * 16 + l16;             // B-matrix lane = n = j
#pragma unroll
        for (int vv = 0; vv < 8; ++vv) {
          int hd0 = ((vv & 4) << 2) + hsel * 8 + ((vv & 3) << 1);    // K pair, table order
          kf0.u[vv] = *(const unsigned*)&k_lds[jrow * 1024 + h * HDd + hd0];
          kf1.u[vv] = *(const unsigned*)&k_lds[jrow * 1024 + h * HDd + 32 + hd0];
        }
        v8f c = 0.0f;
        c = wmma_bf16(qf[hh][0], kf0, c);
        c = wmma_bf16(qf[hh][1], kf1, c);
        int mv = mask[j0 + jt * 16 + l16];
#pragma unroll
        for (int r = 0; r < 8; ++r) {
          float s = c[r] * SCALEf;
          if (mv == 0) s = -1e9f;
          int il = r + 8 * hsel;
          sc[(il * 32 + jt * 16 + l16) * 16 + h] = s;
        }
      }
    }
    __syncthreads();

    // softmax over heads: 512 (i,j) positions / 256 threads
#pragma unroll
    for (int r = 0; r < 2; ++r) {
      int t  = tid + r * 256;
      int il = t >> 5, jl = t & 31;
      const float* sp = &sc[(il * 32 + jl) * 16];
      float mx = sp[0];
#pragma unroll
      for (int h = 1; h < Hh; ++h) mx = fmaxf(mx, sp[h]);
      float e[Hh], sum = 0.f;
#pragma unroll
      for (int h = 0; h < Hh; ++h) { e[h] = __expf(sp[h] - mx); sum += e[h]; }
      float inv = 1.0f / sum;
#pragma unroll
      for (int h = 0; h < Hh; ++h)
        at[(h * 16 + il) * 32 + jl] = f2bf(e[h] * inv);
    }
    __syncthreads();

    // attn @ V: A = attn (16 i x 32 j), B = v (32 j x 16 hd), 4 hd-subtiles
#pragma unroll
    for (int hh = 0; hh < 2; ++hh) {
      int h = w8 * 2 + hh;
      BFrag af;
#pragma unroll
      for (int vv = 0; vv < 8; ++vv) {
        int jj = ((vv & 4) << 2) + hsel * 8 + ((vv & 3) << 1);
        af.u[vv] = *(const unsigned*)&at[(h * 16 + l16) * 32 + jj];
      }
#pragma unroll
      for (int nt = 0; nt < 4; ++nt) {
        BFrag vf;
        int d = h * HDd + nt * 16 + l16;
#pragma unroll
        for (int vv = 0; vv < 8; ++vv) {
          int jj = ((vv & 4) << 2) + hsel * 8 + ((vv & 3) << 1);
          vf.u[vv] = *(const unsigned*)&v_lds[d * 32 + jj];
        }
        oacc[hh][nt] = wmma_bf16(af, vf, oacc[hh][nt]);
      }
    }
    __syncthreads();
  }

  // write fp32 context [B,S,D]
#pragma unroll
  for (int hh = 0; hh < 2; ++hh) {
    int h = w8 * 2 + hh;
#pragma unroll
    for (int nt = 0; nt < 4; ++nt) {
      int gd = h * HDd + nt * 16 + l16;
#pragma unroll
      for (int r = 0; r < 8; ++r) {
        int gi = i0 + r + 8 * hsel;
        ctx[(size_t)(b * Ss + gi) * Dd + gd] = oacc[hh][nt][r];
      }
    }
  }
}

// ---------------------------------------------------------------------------
extern "C" void kernel_launch(void* const* d_in, const int* in_sizes, int n_in,
                              void* d_out, int out_size, void* d_ws, size_t ws_size,
                              hipStream_t stream) {
  (void)in_sizes; (void)n_in; (void)out_size; (void)ws_size;
  const float* query = (const float*)d_in[0];
  const float* key   = (const float*)d_in[1];
  const float* value = (const float*)d_in[2];
  const int*   mask  = (const int*)d_in[3];
  const float* Wq = (const float*)d_in[4];  const float* bq = (const float*)d_in[5];
  const float* Wk = (const float*)d_in[6];  const float* bk = (const float*)d_in[7];
  const float* Wv = (const float*)d_in[8];  const float* bv = (const float*)d_in[9];
  const float* Wo = (const float*)d_in[10]; const float* bo = (const float*)d_in[11];

  const size_t NEL = (size_t)Bb * Ss * Dd;          // 4,194,304
  unsigned short* qb = (unsigned short*)d_ws;       // bf16 q
  unsigned short* kb = qb + NEL;                    // bf16 k
  unsigned short* vb = kb + NEL;                    // bf16 v
  float*          cb = (float*)(vb + NEL);          // fp32 context

  dim3 grid(Dd / 128, (Bb * Ss) / 128);             // (8, 32)
  dim3 block(256);
  gemm_bias_kernel<false><<<grid, block, 0, stream>>>(query, Wq, bq, qb);
  gemm_bias_kernel<false><<<grid, block, 0, stream>>>(key,   Wk, bk, kb);
  gemm_bias_kernel<false><<<grid, block, 0, stream>>>(value, Wv, bv, vb);

  const size_t smem = 32 * 1024 * 2 + 1024 * 32 * 2 + 16 * 32 * 16 * 4 +
                      16 * 16 * 32 * 2;             // 180,224 bytes
  hipFuncSetAttribute((const void*)attn_kernel,
                      hipFuncAttributeMaxDynamicSharedMemorySize, (int)smem);
  attn_kernel<<<Bb * (Ss / 16), block, smem, stream>>>(qb, kb, vb, mask, cb);

  gemm_bias_kernel<true><<<grid, block, 0, stream>>>(cb, Wo, bo, (float*)d_out);
}
